// GIN3_57071525429597
// MI455X (gfx1250) — matrix operand order repaired and checked by reading, not compiled, not run on hardware
//
#include <hip/hip_runtime.h>
#include <hip/hip_bf16.h>

typedef __attribute__((ext_vector_type(16))) _Float16 v16h;
typedef __attribute__((ext_vector_type(8)))  float    v8f;

#define NN 50000
#define EE 800000
#define HH 128
#define LL 3
#define CC 10
#define GG 512

// ---------------------------------------------------------------------------
// CDNA5 async global->LDS copy (ASYNCcnt-tracked, no VGPR staging).
// vdst operand = LDS byte address (low 32 bits of the generic pointer),
// vaddr = 64-bit global address, saddr = off.
// ---------------------------------------------------------------------------
__device__ __forceinline__ void async_g2l_b128(unsigned int lds_addr,
                                               const void* gaddr) {
    asm volatile("global_load_async_to_lds_b128 %0, %1, off"
                 :: "v"(lds_addr), "v"(gaddr)
                 : "memory");
}
__device__ __forceinline__ void wait_async0() {
    asm volatile("s_wait_asynccnt 0" ::: "memory");
}

// ---------------------------------------------------------------------------
// Build B-matrix fragments for V_WMMA_F32_16X16X32_F16 from a row-major
// K(=128) x N(=128) fp32 weight matrix.
// Fragment layout: frag[((kb*8 + nt)*32 + lane)*16 + j]  (halves)
//   lane&15 -> column within 16-wide N tile, lane&16 -> +8 on K,
//   element j<8 -> K offset j, j>=8 -> K offset 16+(j-8).
// ---------------------------------------------------------------------------
__global__ void build_wfrag_kernel(const float* __restrict__ W,
                                   _Float16* __restrict__ frag) {
    int t = blockIdx.x * blockDim.x + threadIdx.x;
    if (t >= 4 * 8 * 32) return;
    int lane = t & 31;
    int nt   = (t >> 5) & 7;
    int kb   = t >> 8;                       // 0..3
    int col  = nt * 16 + (lane & 15);
    int krow = kb * 32 + ((lane & 16) ? 8 : 0);
    _Float16* dst = frag + ((size_t)((kb * 8 + nt) * 32 + lane)) * 16;
#pragma unroll
    for (int j = 0; j < 16; ++j)
        dst[j] = (_Float16)W[(krow + (j < 8 ? j : 8 + j)) * HH + col];
}

// ---------------------------------------------------------------------------
// GIN aggregation: one wave32 per edge, each lane handles 4 channels.
// agg[dst] += h[src] via fp32 L2 atomics (h & agg are L2-resident: 25.6MB).
// Edge indices forced wave-uniform -> scalar loads.
// ---------------------------------------------------------------------------
__global__ void scatter_agg_kernel(const float* __restrict__ h,
                                   const int* __restrict__ src,
                                   const int* __restrict__ dst,
                                   float* __restrict__ agg, int nEdges) {
    int t = blockIdx.x * blockDim.x + threadIdx.x;
    int e = __builtin_amdgcn_readfirstlane(t >> 5);
    int lane = threadIdx.x & 31;
    if (e >= nEdges) return;
    int s = __builtin_amdgcn_readfirstlane(src[e]);
    int d = __builtin_amdgcn_readfirstlane(dst[e]);
    const float4 v = *(const float4*)(h + (size_t)s * HH + lane * 4);
    float* ap = agg + (size_t)d * HH + lane * 4;
    atomicAdd(ap + 0, v.x);
    atomicAdd(ap + 1, v.y);
    atomicAdd(ap + 2, v.z);
    atomicAdd(ap + 3, v.w);
}

// ---------------------------------------------------------------------------
// Mean-pool accumulation: one wave per node.
// ---------------------------------------------------------------------------
__global__ void pool_kernel(const float* __restrict__ h,
                            const int* __restrict__ batch,
                            float* __restrict__ pooled,
                            float* __restrict__ counts, int nNodes) {
    int t = blockIdx.x * blockDim.x + threadIdx.x;
    int node = __builtin_amdgcn_readfirstlane(t >> 5);
    int lane = threadIdx.x & 31;
    if (node >= nNodes) return;
    int b = __builtin_amdgcn_readfirstlane(batch[node]);
    if (lane == 0) atomicAdd(&counts[b], 1.0f);
    const float4 v = *(const float4*)(h + (size_t)node * HH + lane * 4);
    float* p = pooled + (size_t)b * HH + lane * 4;
    atomicAdd(p + 0, v.x);
    atomicAdd(p + 1, v.y);
    atomicAdd(p + 2, v.z);
    atomicAdd(p + 3, v.w);
}

// ---------------------------------------------------------------------------
// WMMA GEMM: out[M x 128] = epilogue( A[M x 128] @ W[128 x 128] )
// Weight fragments staged global->LDS with async copies (once per block),
// then each wave owns a 16-row strip; per K-step all 8 B fragments are
// batch-loaded from LDS before the 8 WMMAs so ds latency overlaps issue.
// modeA: 0 = A0 plain; 1 = (1+eps)*A0 + A1 (GIN fuse); 2 = A0 / max(counts,1)
// modeE: 0 = relu(x + bias); 1 = BN( relu(x + bias) )
// ---------------------------------------------------------------------------
__global__ __launch_bounds__(256) void wmma_gemm128_kernel(
    const float* __restrict__ A0, const float* __restrict__ A1,
    const float* __restrict__ epsp, const float* __restrict__ countsp,
    const _Float16* __restrict__ wfrag, const float* __restrict__ bias,
    const float* __restrict__ gamma, const float* __restrict__ beta,
    const float* __restrict__ mean, const float* __restrict__ var,
    float* __restrict__ out, int Mtiles, int modeA, int modeE) {
    __shared__ _Float16 lds_w[HH * HH];      // 32 KB weight fragments

    // ---- async stage: 32768 B / 256 threads = 8 x b128 per thread ----
    {
        unsigned int lbase = (unsigned int)(unsigned long long)(&lds_w[0]);
        const char* gsrc = (const char*)wfrag;
#pragma unroll
        for (int i = 0; i < 8; ++i) {
            unsigned int boff = (threadIdx.x + i * 256) * 16;
            async_g2l_b128(lbase + boff, gsrc + boff);
        }
        wait_async0();
        __syncthreads();
    }

    int wave = (blockIdx.x * blockDim.x + threadIdx.x) >> 5;
    int lane = threadIdx.x & 31;
    if (wave >= Mtiles) return;              // whole-wave uniform exit

    int m0 = wave * 16;
    int arow = m0 + (lane & 15);             // A row handled by this lane
    float ope  = (modeA == 1) ? (1.0f + epsp[0]) : 1.0f;
    float invc = 1.0f;
    if (modeA == 2) invc = 1.0f / fmaxf(countsp[arow], 1.0f);

    const v16h* wf = (const v16h*)lds_w;
    v8f acc[8] = {};

#pragma unroll
    for (int kb = 0; kb < 4; ++kb) {
        // ---- A fragment (16x32 f16, ISA layout) ----
        int kbase = kb * 32 + ((lane & 16) ? 8 : 0);
        const float* a0r = A0 + (size_t)arow * HH + kbase;
        v16h a;
        if (modeA == 1) {
            const float* a1r = A1 + (size_t)arow * HH + kbase;
#pragma unroll
            for (int j = 0; j < 8; ++j) {
                a[j]     = (_Float16)(ope * a0r[j]      + a1r[j]);
                a[8 + j] = (_Float16)(ope * a0r[16 + j] + a1r[16 + j]);
            }
        } else if (modeA == 2) {
#pragma unroll
            for (int j = 0; j < 8; ++j) {
                a[j]     = (_Float16)(a0r[j]      * invc);
                a[8 + j] = (_Float16)(a0r[16 + j] * invc);
            }
        } else {
#pragma unroll
            for (int j = 0; j < 8; ++j) {
                a[j]     = (_Float16)a0r[j];
                a[8 + j] = (_Float16)a0r[16 + j];
            }
        }
        // ---- batch all 8 B-fragment LDS loads, then 8 WMMAs ----
        v16h bf[8];
#pragma unroll
        for (int nt = 0; nt < 8; ++nt)
            bf[nt] = wf[(kb * 8 + nt) * 32 + lane];
#pragma unroll
        for (int nt = 0; nt < 8; ++nt)
            acc[nt] = __builtin_amdgcn_wmma_f32_16x16x32_f16(
                false, a, false, bf[nt], (short)0, acc[nt], false, false);
    }

    // ---- epilogue + store (D layout: vgpr r -> row r / r+8) ----
    int col16 = lane & 15;
    int rowbase = m0 + ((lane & 16) ? 8 : 0);
#pragma unroll
    for (int nt = 0; nt < 8; ++nt) {
        int col = nt * 16 + col16;
        float bb = bias[col];
        float sc = 1.0f, sh = 0.0f;
        if (modeE == 1) {
            sc = gamma[col] * rsqrtf(var[col] + 1e-5f);
            sh = beta[col] - mean[col] * sc;
        }
#pragma unroll
        for (int r = 0; r < 8; ++r) {
            float v = acc[nt][r] + bb;
            v = fmaxf(v, 0.0f);
            if (modeE == 1) v = v * sc + sh;
            out[(size_t)(rowbase + r) * HH + col] = v;
        }
    }
}

// ---------------------------------------------------------------------------
// Head: logits = g @ lin2_w + lin2_b; out = log_softmax(logits).
// One thread per graph (512 threads, 10x128 MACs each — negligible).
// ---------------------------------------------------------------------------
__global__ void head_kernel(const float* __restrict__ g,
                            const float* __restrict__ w2,
                            const float* __restrict__ b2,
                            float* __restrict__ out) {
    int gi = blockIdx.x * blockDim.x + threadIdx.x;
    if (gi >= GG) return;
    const float* gr = g + (size_t)gi * HH;
    float logit[CC];
    float mx = -3.402823466e+38f;
#pragma unroll
    for (int c = 0; c < CC; ++c) {
        float acc = b2[c];
        for (int k = 0; k < HH; ++k) acc += gr[k] * w2[k * CC + c];
        logit[c] = acc;
        mx = fmaxf(mx, acc);
    }
    float s = 0.0f;
#pragma unroll
    for (int c = 0; c < CC; ++c) s += __expf(logit[c] - mx);
    float lse = mx + __logf(s);
#pragma unroll
    for (int c = 0; c < CC; ++c) out[gi * CC + c] = logit[c] - lse;
}

// ---------------------------------------------------------------------------
extern "C" void kernel_launch(void* const* d_in, const int* in_sizes, int n_in,
                              void* d_out, int out_size, void* d_ws, size_t ws_size,
                              hipStream_t stream) {
    const float* x       = (const float*)d_in[0];
    const int*   eidx    = (const int*)d_in[1];      // (2,E): [0:E]=src, [E:2E]=dst
    const int*   batch   = (const int*)d_in[2];
    const float* W1      = (const float*)d_in[3];    // (L,H,H)
    const float* b1      = (const float*)d_in[4];
    const float* W2      = (const float*)d_in[5];
    const float* b2      = (const float*)d_in[6];
    const float* gamma   = (const float*)d_in[7];
    const float* beta    = (const float*)d_in[8];
    const float* rmean   = (const float*)d_in[9];
    const float* rvar    = (const float*)d_in[10];
    const float* eps_gin = (const float*)d_in[11];
    const float* lin1_w  = (const float*)d_in[12];
    const float* lin1_b  = (const float*)d_in[13];
    const float* lin2_w  = (const float*)d_in[14];
    const float* lin2_b  = (const float*)d_in[15];
    float* out = (float*)d_out;

    // ---- workspace carve-up (all 256B aligned) ----
    char* ws = (char*)d_ws;
    size_t off = 0;
    auto carve = [&](size_t bytes) -> char* {
        char* p = ws + off;
        off = (off + bytes + 255) & ~(size_t)255;
        return p;
    };
    const size_t NH = (size_t)NN * HH;
    float*    h      = (float*)carve(NH * 4);
    float*    agg    = (float*)carve(NH * 4);
    float*    z1     = (float*)carve(NH * 4);
    _Float16* fW1    = (_Float16*)carve((size_t)LL * HH * HH * 2);
    _Float16* fW2    = (_Float16*)carve((size_t)LL * HH * HH * 2);
    _Float16* fL1    = (_Float16*)carve((size_t)HH * HH * 2);
    float*    pooled = (float*)carve((size_t)GG * HH * 4);
    float*    counts = (float*)carve((size_t)GG * 4);
    float*    gbuf   = (float*)carve((size_t)GG * HH * 4);
    (void)ws_size; (void)n_in; (void)in_sizes; (void)out_size;

    // ---- stage weights into WMMA B-fragment layout (once per launch) ----
    for (int l = 0; l < LL; ++l) {
        build_wfrag_kernel<<<4, 256, 0, stream>>>(W1 + (size_t)l * HH * HH,
                                                  fW1 + (size_t)l * HH * HH);
        build_wfrag_kernel<<<4, 256, 0, stream>>>(W2 + (size_t)l * HH * HH,
                                                  fW2 + (size_t)l * HH * HH);
    }
    build_wfrag_kernel<<<4, 256, 0, stream>>>(lin1_w, fL1);

    // h = x (do not mutate inputs)
    hipMemcpyAsync(h, x, NH * 4, hipMemcpyDeviceToDevice, stream);

    const int Mtiles   = NN / 16;                       // 3125 (exact)
    const int gemmGrid = (Mtiles + 7) / 8;              // 8 waves per block
    const int scatGrid = (int)(((size_t)EE * 32 + 255) / 256);
    const int poolGrid = (int)(((size_t)NN * 32 + 255) / 256);

    for (int l = 0; l < LL; ++l) {
        hipMemsetAsync(agg, 0, NH * 4, stream);
        scatter_agg_kernel<<<scatGrid, 256, 0, stream>>>(h, eidx, eidx + EE, agg, EE);
        // z1 = relu( ((1+eps)*h + agg) @ W1 + b1 )
        wmma_gemm128_kernel<<<gemmGrid, 256, 0, stream>>>(
            h, agg, eps_gin + l, nullptr, fW1 + (size_t)l * HH * HH,
            b1 + (size_t)l * HH, nullptr, nullptr, nullptr, nullptr,
            z1, Mtiles, /*modeA=*/1, /*modeE=*/0);
        // h = BN( relu( z1 @ W2 + b2 ) )
        wmma_gemm128_kernel<<<gemmGrid, 256, 0, stream>>>(
            z1, nullptr, nullptr, nullptr, fW2 + (size_t)l * HH * HH,
            b2 + (size_t)l * HH, gamma + (size_t)l * HH, beta + (size_t)l * HH,
            rmean + (size_t)l * HH, rvar + (size_t)l * HH,
            h, Mtiles, /*modeA=*/0, /*modeE=*/1);
    }

    // ---- global mean pool + head ----
    hipMemsetAsync(pooled, 0, (size_t)GG * HH * 4, stream);
    hipMemsetAsync(counts, 0, (size_t)GG * 4, stream);
    pool_kernel<<<poolGrid, 256, 0, stream>>>(h, batch, pooled, counts, NN);

    // g = relu( (pooled / counts) @ lin1_w + lin1_b )   (32 M-tiles)
    wmma_gemm128_kernel<<<(32 + 7) / 8, 256, 0, stream>>>(
        pooled, nullptr, nullptr, counts, fL1, lin1_b,
        nullptr, nullptr, nullptr, nullptr,
        gbuf, 32, /*modeA=*/2, /*modeE=*/0);

    head_kernel<<<(GG + 255) / 256, 256, 0, stream>>>(gbuf, lin2_w, lin2_b, out);
}
